// SimpleRawModel_29214367547833
// MI455X (gfx1250) — 16-bit and 8-bit matrix operand order repaired and checked by reading, compiled, not run on hardware
//
#include <hip/hip_runtime.h>


typedef __attribute__((ext_vector_type(16))) int      v16i;
typedef __attribute__((ext_vector_type(4)))  int      v4i;
typedef __attribute__((ext_vector_type(8)))  float    v8f;
typedef __attribute__((ext_vector_type(4)))  float    v4f;
typedef __attribute__((ext_vector_type(16))) _Float16 v16h;
typedef __attribute__((ext_vector_type(8)))  _Float16 v8h;
typedef __attribute__((ext_vector_type(4)))  _Float16 v4h;

#define N_USERS 20000
#define N_ITEMS 8192
#define BATCH   2048
#define KPAD    20096
#define NKCH    (KPAD / 128)
#define FP8_ONE 0x38

#if __has_builtin(__builtin_amdgcn_global_load_async_to_lds_b128)
#define HAVE_ASYNC_LDS 1
#else
#define HAVE_ASYNC_LDS 0
#endif

#if HAVE_ASYNC_LDS
#if __has_builtin(__builtin_amdgcn_s_wait_asynccnt)
#define WAIT_ASYNC(n) __builtin_amdgcn_s_wait_asynccnt(n)
#else
#define WAIT_ASYNC(n) asm volatile("s_wait_asynccnt %0" ::"i"(n) : "memory")
#endif

typedef __attribute__((address_space(1))) v4i glb_v4i;
typedef __attribute__((address_space(3))) v4i lds_v4i;

__device__ __forceinline__ void async_copy16(const unsigned char* g, unsigned char* l)
{
    __builtin_amdgcn_global_load_async_to_lds_b128(
        (glb_v4i*)g, (lds_v4i*)l, 0, 0);
}
#endif

__global__ __launch_bounds__(256) void k_quant_transpose(
    const float* __restrict__ adj, unsigned char* __restrict__ adjT)
{
    __shared__ unsigned char tile[64][80];
    const int u0 = blockIdx.x * 64;
    const int i0 = blockIdx.y * 64;
    const int t  = threadIdx.x;
    const int r  = t >> 2;
    const int cq = t & 3;
    const int u  = u0 + r;
    if (u < N_USERS) {
        const float* src = adj + (long)u * N_ITEMS + i0 + cq * 16;
        #pragma unroll
        for (int k = 0; k < 4; ++k) {
            v4f x = *(const v4f*)(src + k * 4);
            #pragma unroll
            for (int e = 0; e < 4; ++e)
                tile[cq * 16 + k * 4 + e][r] = (x[e] != 0.0f) ? FP8_ONE : 0;
        }
    } else {
        #pragma unroll
        for (int k = 0; k < 16; ++k) tile[cq * 16 + k][r] = 0;
    }
    __syncthreads();
    const int il = t >> 2;
    const int ch = t & 3;
    v4i val = *(const v4i*)&tile[il][ch * 16];
    *(v4i*)(adjT + (long)(i0 + il) * KPAD + u0 + ch * 16) = val;
}

__global__ __launch_bounds__(256) void k_gather_batch(
    const float* __restrict__ adj, const int* __restrict__ users,
    _Float16* __restrict__ batch)
{
    const int b = blockIdx.x;
    const int u = users[b];
    const float* src = adj + (long)u * N_ITEMS;
    _Float16* dst    = batch + (long)b * N_ITEMS;
    for (int j = threadIdx.x * 4; j < N_ITEMS; j += 256 * 4) {
        v4f x = *(const v4f*)(src + j);
        v4h h = { (_Float16)x[0], (_Float16)x[1], (_Float16)x[2], (_Float16)x[3] };
        *(v4h*)(dst + j) = h;
    }
}

__device__ __forceinline__ void cooc_compute_chunk(
    const unsigned char* __restrict__ sA, const unsigned char* __restrict__ sB,
    int w, int lane, int hi, v8f acc[8])
{
    union { v16i v; unsigned long long d[8]; } a;
    const unsigned char* arow = sA + (w * 16 + (lane & 15)) * 128;
    const int hofA = hi ? 16 : 0;
    #pragma unroll
    for (int k = 0; k < 8; ++k)
        a.d[k] = *(const unsigned long long*)(arow + 32 * (k >> 1) + 8 * (k & 1) + hofA);

    #pragma unroll
    for (int j = 0; j < 8; ++j) {
        union { v16i v; v4i q[4]; } bfrag;
        const unsigned char* brow = sB + (j * 16 + (lane & 15)) * 128;
        const int hofB = hi ? 16 : 0;
        #pragma unroll
        for (int k = 0; k < 4; ++k)
            bfrag.q[k] = *(const v4i*)(brow + 32 * k + hofB);
        acc[j] = __builtin_amdgcn_wmma_f32_16x16x128_fp8_fp8(
            a.v, bfrag.v, (short)0, acc[j], false, false);
    }
}

__global__ __launch_bounds__(256) void k_cooc_gemm(
    const unsigned char* __restrict__ adjT, float* __restrict__ sim)
{
    const int bi = blockIdx.x, bj = blockIdx.y;
    const int t = threadIdx.x;
    const int lane = t & 31, w = t >> 5, hi = lane >> 4;
    v8f acc[8] = {};

    const unsigned char* gA = adjT + (long)(bi * 128) * KPAD;
    const unsigned char* gB = adjT + (long)(bj * 128) * KPAD;

#if HAVE_ASYNC_LDS
    __shared__ unsigned char sA[2][128 * 128];
    __shared__ unsigned char sB[2][128 * 128];

    #pragma unroll
    for (int s = 0; s < 4; ++s) {
        const int c = t + s * 256;
        const int row = c >> 3, off = (c & 7) * 16;
        async_copy16(gA + (long)row * KPAD + off, &sA[0][row * 128 + off]);
        async_copy16(gB + (long)row * KPAD + off, &sB[0][row * 128 + off]);
    }

    for (int kc = 0; kc < NKCH; ++kc) {
        const int cur = kc & 1;
        if (kc + 1 < NKCH) {
            #pragma unroll
            for (int s = 0; s < 4; ++s) {
                const int c = t + s * 256;
                const int row = c >> 3, off = (c & 7) * 16;
                async_copy16(gA + (long)row * KPAD + (kc + 1) * 128 + off,
                             &sA[cur ^ 1][row * 128 + off]);
                async_copy16(gB + (long)row * KPAD + (kc + 1) * 128 + off,
                             &sB[cur ^ 1][row * 128 + off]);
            }
            WAIT_ASYNC(8);
        } else {
            WAIT_ASYNC(0);
        }
        __syncthreads();
        cooc_compute_chunk(sA[cur], sB[cur], w, lane, hi, acc);
        __syncthreads();
    }
#else
    __shared__ unsigned char sA[128 * 128];
    __shared__ unsigned char sB[128 * 128];

    for (int kc = 0; kc < NKCH; ++kc) {
        #pragma unroll
        for (int s = 0; s < 4; ++s) {
            const int c = t + s * 256;
            const int row = c >> 3, off = (c & 7) * 16;
            *(v4i*)(sA + row * 128 + off) =
                *(const v4i*)(gA + (long)row * KPAD + kc * 128 + off);
            *(v4i*)(sB + row * 128 + off) =
                *(const v4i*)(gB + (long)row * KPAD + kc * 128 + off);
        }
        __syncthreads();
        if (kc + 1 < NKCH) {
            __builtin_prefetch(gA + (long)(t >> 1) * KPAD + (kc + 1) * 128 + (t & 1) * 64, 0, 1);
            __builtin_prefetch(gB + (long)(t >> 1) * KPAD + (kc + 1) * 128 + (t & 1) * 64, 0, 1);
        }
        cooc_compute_chunk(sA, sB, w, lane, hi, acc);
        __syncthreads();
    }
#endif

    const int col = lane & 15;
    const long gm = (long)bi * 128 + w * 16 + hi * 8;
    const long gn = (long)bj * 128 + col;
    #pragma unroll
    for (int j = 0; j < 8; ++j)
        #pragma unroll
        for (int v = 0; v < 8; ++v)
            sim[(gm + v) * N_ITEMS + gn + j * 16] = acc[j][v];
}

__global__ __launch_bounds__(256) void k_softmax_norm(
    float* __restrict__ sim, const float* __restrict__ temperature,
    _Float16* __restrict__ norm)
{
    __shared__ float red[256];
    const int row = blockIdx.x;
    const int t = threadIdx.x;
    const float invT = 1.0f / temperature[0];
    float* x = sim + (long)row * N_ITEMS;

    float m = -3.402823466e+38f;
    for (int j = t; j < N_ITEMS; j += 256) m = fmaxf(m, x[j] * invT);
    red[t] = m;
    __syncthreads();
    for (int s = 128; s > 0; s >>= 1) {
        if (t < s) red[t] = fmaxf(red[t], red[t + s]);
        __syncthreads();
    }
    m = red[0];
    __syncthreads();

    float sum = 0.0f;
    for (int j = t; j < N_ITEMS; j += 256) {
        const float e = __expf(x[j] * invT - m);
        x[j] = e;
        sum += e;
    }
    red[t] = sum;
    __syncthreads();
    for (int s = 128; s > 0; s >>= 1) {
        if (t < s) red[t] += red[t + s];
        __syncthreads();
    }
    const float scale = 1.0f / (red[0] + 1e-10f);

    _Float16* o = norm + (long)row * N_ITEMS;
    for (int j = t; j < N_ITEMS; j += 256) o[j] = (_Float16)(x[j] * scale);
}

__global__ __launch_bounds__(256) void k_score_gemm(
    const _Float16* __restrict__ batch, const _Float16* __restrict__ norm,
    float* __restrict__ out)
{
    __shared__ _Float16 sAh[128][64];
    __shared__ _Float16 sBT[128][72];
    const int bm = blockIdx.x, bn = blockIdx.y;
    const int t = threadIdx.x;
    const int lane = t & 31, w = t >> 5, hi = lane >> 4;
    v8f acc[8] = {};

    for (int kc = 0; kc < N_ITEMS / 64; ++kc) {
        #pragma unroll
        for (int s = 0; s < 4; ++s) {
            const int c = t + s * 256;
            const int row = c >> 3, off = (c & 7) * 8;
            *(v4i*)(&sAh[row][off]) =
                *(const v4i*)(batch + (long)(bm * 128 + row) * N_ITEMS + kc * 64 + off);
        }
        #pragma unroll
        for (int s = 0; s < 4; ++s) {
            const int c = t + s * 256;
            const int kk = c >> 4, noff = (c & 15) * 8;
            v8h vsrc = *(const v8h*)(norm + (long)(kc * 64 + kk) * N_ITEMS + bn * 128 + noff);
            #pragma unroll
            for (int e = 0; e < 8; ++e) sBT[noff + e][kk] = vsrc[e];
        }
        __syncthreads();

        #pragma unroll
        for (int ks = 0; ks < 2; ++ks) {
            union { v16h v; v4i q[2]; } a;
            const unsigned char* arow =
                (const unsigned char*)&sAh[w * 16 + (lane & 15)][0] + ks * 64;
            a.q[0] = *(const v4i*)(arow + (hi ? 16 : 0));
            a.q[1] = *(const v4i*)(arow + 32 + (hi ? 16 : 0));
            #pragma unroll
            for (int j = 0; j < 8; ++j) {
                union { v16h v; v4i q[2]; } b;
                const unsigned char* brow =
                    (const unsigned char*)&sBT[j * 16 + (lane & 15)][0] + ks * 64;
                b.q[0] = *(const v4i*)(brow + (hi ? 16 : 0));
                b.q[1] = *(const v4i*)(brow + 32 + (hi ? 16 : 0));
                acc[j] = __builtin_amdgcn_wmma_f32_16x16x32_f16(
                    false, a.v, false, b.v, (short)0, acc[j], false, false);
            }
        }
        __syncthreads();
    }

    const int col = lane & 15;
    const long gm = (long)bm * 128 + w * 16 + hi * 8;
    const long gn = (long)bn * 128 + col;
    #pragma unroll
    for (int j = 0; j < 8; ++j)
        #pragma unroll
        for (int v = 0; v < 8; ++v)
            out[(gm + v) * N_ITEMS + gn + j * 16] = acc[j][v];
}

extern "C" void kernel_launch(void* const* d_in, const int* in_sizes, int n_in,
                              void* d_out, int out_size, void* d_ws, size_t ws_size,
                              hipStream_t stream)
{
    const float* adj         = (const float*)d_in[0];
    const float* temperature = (const float*)d_in[1];
    const int*   users       = (const int*)  d_in[2];
    float*       out         = (float*)d_out;
    char*        ws          = (char*)d_ws;

    unsigned char* adjT  = (unsigned char*)ws;
    float*         sim   = (float*)(ws + 164626432L);
    _Float16*      batch = (_Float16*)(ws + 164626432L + 268435456L);
    _Float16*      norm  = (_Float16*)ws;

    k_quant_transpose<<<dim3(KPAD / 64, N_ITEMS / 64), 256, 0, stream>>>(adj, adjT);
    k_gather_batch  <<<dim3(BATCH),                    256, 0, stream>>>(adj, users, batch);
    k_cooc_gemm     <<<dim3(N_ITEMS / 128, N_ITEMS / 128), 256, 0, stream>>>(adjT, sim);
    k_softmax_norm  <<<dim3(N_ITEMS),                  256, 0, stream>>>(sim, temperature, norm);
    k_score_gemm    <<<dim3(BATCH / 128, N_ITEMS / 128), 256, 0, stream>>>(batch, norm, out);
}
